// TFC_LaSAFT_66984309948466
// MI455X (gfx1250) — matrix operand order repaired and checked by reading, not compiled, not run on hardware
//
#include <hip/hip_runtime.h>

// ---------------- types ----------------
typedef __bf16 bf16;
typedef __attribute__((ext_vector_type(16))) bf16  v16bf;
typedef __attribute__((ext_vector_type(8)))  bf16  bf16x8;
typedef __attribute__((ext_vector_type(2)))  bf16  v2bf;
typedef __attribute__((ext_vector_type(8)))  float v8f;

#define GN_EPS 1e-5f
#define LSTR 40  // LDS tile row stride in halves (80B, 16B-aligned, bank-skewed)

// pack two f32 into 2xbf16 (RNE; lowers to packed cvt where available)
__device__ __forceinline__ unsigned packbf(float lo, float hi) {
  v2bf r;
  r[0] = (bf16)lo;
  r[1] = (bf16)hi;
  return __builtin_bit_cast(unsigned, r);
}

// ---------------- WMMA fragment loads (CDNA5 16x16x32 bf16 layout) ----------------
// A tile in LDS: [16 rows][LSTR halves], K contiguous. Lane L: m=L&15,
// halves 0..7 = K (L>=16?8:0)+0..7 ; halves 8..15 = same +16.
__device__ __forceinline__ v16bf load_a_frag(const bf16* As, int lane) {
  int m  = lane & 15;
  int kb = (lane & 16) ? 8 : 0;
  const bf16* p = As + m * LSTR + kb;
  bf16x8 lo = *(const bf16x8*)(p);
  bf16x8 hi = *(const bf16x8*)(p + 16);
  return __builtin_shufflevector(lo, hi, 0,1,2,3,4,5,6,7,8,9,10,11,12,13,14,15);
}
// B tile in LDS stored N-major: [16 N-rows][LSTR halves of K]. Lane L: n=L&15,
// halves 0..7 = K (L>=16?8:0)+0..7 ; halves 8..15 = same +16.
__device__ __forceinline__ v16bf load_b_frag(const bf16* Bs, int lane) {
  int n  = lane & 15;
  int kb = (lane & 16) ? 8 : 0;
  const bf16* p = Bs + n * LSTR + kb;
  bf16x8 lo = *(const bf16x8*)(p);
  bf16x8 hi = *(const bf16x8*)(p + 16);
  return __builtin_shufflevector(lo, hi, 0,1,2,3,4,5,6,7,8,9,10,11,12,13,14,15);
}
__device__ __forceinline__ v8f wmma_bf16(v16bf a, v16bf b, v8f c) {
  return __builtin_amdgcn_wmma_f32_16x16x32_bf16(false, a, false, b, (short)0, c, false, false);
}

// ---------------- conv 3x3 (implicit GEMM, bf16 WMMA, double-buffered) -----------
// Y[b, n<24, t, f] = conv(src, W) + bias ; GN sum/sumsq per (b,group) fused into
// ostats. Dense-concat inputs read from up to 3 buffers (24ch each).
// Block: 128 threads (4 waves), tile = 64 f-positions x 32 N (24 used).
template<int CIN>
__global__ void __launch_bounds__(128)
conv3x3_wmma(const float* __restrict__ s0, const float* __restrict__ s1,
             const float* __restrict__ s2,
             const float* __restrict__ W, const float* __restrict__ bias,
             float* __restrict__ Y, float* __restrict__ ostats) {
  constexpr int K   = CIN * 9;
  constexpr int KP  = (K + 31) & ~31;
  constexpr int NST = KP / 32;
  __shared__ __align__(16) bf16 As[2][64 * LSTR];
  __shared__ __align__(16) bf16 Bs[2][32 * LSTR];
  __shared__ float gsum[4], gsq[4];

  const int tid  = threadIdx.x;
  const int lane = tid & 31;
  const int wv   = tid >> 5;                 // 0..3 -> 16 M-rows each
  const int f0   = blockIdx.x * 64;
  const int t    = blockIdx.y;
  const int b    = blockIdx.z;

  if (tid < 4) { gsum[tid] = 0.0f; gsq[tid] = 0.0f; }

  // A fill: thread owns 2 k-columns x 8 consecutive f-rows (contiguous loads)
  const int kk = (tid & 15) * 2;             // k columns kk, kk+1
  const int mc = tid >> 4;                   // rows mc*8 .. mc*8+7
  // B fill: thread owns 8 contiguous k for one n
  const int bn  = tid >> 2;                  // 0..31
  const int bk0 = (tid & 3) * 8;

  v8f acc0 = {}; v8f acc1 = {};
  float af[2][8];                            // raw A prefetch registers
  float wf[8];                               // raw B prefetch registers

  auto fill_raw = [&](int k0) {
    // --- B: W[n][k] is K-contiguous (OIHW, k = ci*9+dt*3+df) ---
    if (bn < 24 && k0 + 32 <= K) {
      const float* wp = W + (size_t)bn * K + k0 + bk0;
#pragma unroll
      for (int i = 0; i < 8; ++i) wf[i] = wp[i];
    } else {
#pragma unroll
      for (int i = 0; i < 8; ++i) {
        int ka = k0 + bk0 + i;
        float v = W[(size_t)(bn < 24 ? bn : 23) * K + (ka < K ? ka : K - 1)];
        wf[i] = (bn < 24 && ka < K) ? v : 0.0f;
      }
    }
    // --- A: branch-free im2col gather (clamped addresses + select) ---
#pragma unroll
    for (int j = 0; j < 2; ++j) {
      int k   = k0 + kk + j;
      int kcl = k < K ? k : K - 1;
      int ci  = kcl / 9;
      int rem = kcl - ci * 9;
      int dt  = rem / 3;
      int df  = rem - dt * 3;
      int tt  = t + dt - 1;
      bool tok = (k < K) && (tt >= 0) && (tt < 128);
      int ttc = tt < 0 ? 0 : (tt > 127 ? 127 : tt);
      const float* sp = (ci < 24) ? s0 : ((ci < 48) ? s1 : s2);
      int cc = (ci < 24) ? ci : ((ci < 48) ? ci - 24 : ci - 48);
      const float* rowp = sp + (((size_t)b * 24 + cc) * 128 + ttc) * 512;
      int fbase = f0 + mc * 8 + df - 1;
#pragma unroll
      for (int i = 0; i < 8; ++i) {
        int ff  = fbase + i;
        int ffc = ff < 0 ? 0 : (ff > 511 ? 511 : ff);
        float v = rowp[ffc];                 // unconditional load
        af[j][i] = (tok && ff >= 0 && ff <= 511) ? v : 0.0f;
      }
    }
  };

  auto store_tiles = [&](int buf) {
#pragma unroll
    for (int i = 0; i < 8; ++i)
      *(unsigned*)(&As[buf][(mc * 8 + i) * LSTR + kk]) = packbf(af[0][i], af[1][i]);
    unsigned q0 = packbf(wf[0], wf[1]), q1 = packbf(wf[2], wf[3]);
    unsigned q2 = packbf(wf[4], wf[5]), q3 = packbf(wf[6], wf[7]);
    *(uint4*)(&Bs[buf][bn * LSTR + bk0]) = make_uint4(q0, q1, q2, q3);
  };

  auto compute = [&](int buf) {
    v16bf a  = load_a_frag(&As[buf][wv * 16 * LSTR], lane);
    v16bf bl = load_b_frag(&Bs[buf][0], lane);
    v16bf bh = load_b_frag(&Bs[buf][16 * LSTR], lane);
    acc0 = wmma_bf16(a, bl, acc0);
    acc1 = wmma_bf16(a, bh, acc1);
  };

  // software pipeline: one barrier per K-step, global loads overlap WMMA
  fill_raw(0);
  store_tiles(0);
#pragma unroll 1
  for (int s = 1; s < NST; ++s) {
    __syncthreads();                // tile s-1 visible; tile s-2 fully consumed
    fill_raw(s * 32);               // long-latency loads for tile s
    compute((s - 1) & 1);           // WMMA on tile s-1
    store_tiles(s & 1);             // pack + stage tile s
  }
  __syncthreads();
  compute((NST - 1) & 1);

  // --- epilogue: store + fused GroupNorm partial stats ---
  const int ncol = lane & 15;
  const int n1   = 16 + ncol;
  const int mhi  = (lane & 16) ? 8 : 0;
  const float bs0 = bias[ncol];
  const float bs1 = (n1 < 24) ? bias[n1] : 0.0f;
  float s0a = 0.0f, q0a = 0.0f, s1a = 0.0f, q1a = 0.0f;
#pragma unroll
  for (int v = 0; v < 8; ++v) {
    int fpos = f0 + wv * 16 + v + mhi;
    float y0 = acc0[v] + bs0;
    Y[(((size_t)b * 24 + ncol) * 128 + t) * 512 + fpos] = y0;
    s0a += y0; q0a += y0 * y0;
    if (n1 < 24) {
      float y1 = acc1[v] + bs1;
      Y[(((size_t)b * 24 + n1) * 128 + t) * 512 + fpos] = y1;
      s1a += y1; q1a += y1 * y1;
    }
  }
  int g0 = ncol / 6;
  atomicAdd(&gsum[g0], s0a); atomicAdd(&gsq[g0], q0a);
  if (n1 < 24) {
    int g1 = n1 / 6;
    atomicAdd(&gsum[g1], s1a); atomicAdd(&gsq[g1], q1a);
  }
  __syncthreads();
  if (tid < 4) {
    atomicAdd(&ostats[(b * 4 + tid) * 2 + 0], gsum[tid]);
    atomicAdd(&ostats[(b * 4 + tid) * 2 + 1], gsq[tid]);
  }
}

// ---------------- dense GEMM: C[M,N] = A[M,K] * B[N,K]^T (bf16 WMMA) -------------
// A_AFF: fused GN+ReLU on A rows (affine constant per row). OUT_BF16: bf16 output.
// OSTATS: fused GN stats of the output (each 64-row block lies in one channel).
// Block: 256 threads = 8 waves (4 M x 2 N), tile 64x64, double-buffered.
template<bool A_AFF, bool OUT_BF16, bool OSTATS>
__global__ void __launch_bounds__(256)
gemm_wmma(const float* __restrict__ A, const float* __restrict__ Bw,
          float* __restrict__ Cf, bf16* __restrict__ Cb,
          int M, int N, int K,
          const float* __restrict__ stats, const float* __restrict__ gw,
          const float* __restrict__ gb, float inv_count,
          float* __restrict__ ostats) {
  __shared__ __align__(16) bf16 As[2][64 * LSTR];
  __shared__ __align__(16) bf16 Bs[2][64 * LSTR];
  __shared__ float sblk, qblk;

  const int tid  = threadIdx.x;
  const int lane = tid & 31;
  const int wv   = tid >> 5;
  const int wm   = wv >> 1;                  // 0..3
  const int wn   = wv & 1;                   // 0..1
  const int m0   = blockIdx.x * 64;
  const int n0   = blockIdx.y * 64;

  if (tid == 0) { sblk = 0.0f; qblk = 0.0f; }

  const int ar   = tid >> 2;                 // fill row 0..63 (A and B)
  const int ak0  = (tid & 3) * 8;
  const int arow = m0 + ar;
  const float* Ap = A  + (size_t)arow * K + ak0;
  const float* Bp = Bw + (size_t)(n0 + ar) * K + ak0;

  float sA = 1.0f, oA = 0.0f;
  if (A_AFF) {                               // GN1 affine: rows are (b, c, t)
    int bb = arow / 3072;
    int cc = (arow % 3072) / 128;
    int g  = cc / 6;
    float sum  = stats[(bb * 4 + g) * 2 + 0];
    float ssq  = stats[(bb * 4 + g) * 2 + 1];
    float mean = sum * inv_count;
    float var  = ssq * inv_count - mean * mean;
    float rs   = rsqrtf(var + GN_EPS);
    sA = gw[cc] * rs;
    oA = gb[cc] - mean * sA;
  }

  v8f acc0 = {}; v8f acc1 = {};
  float afr[8], bfr[8];                      // raw prefetch registers

  auto fill_raw = [&](int k0) {
#pragma unroll
    for (int i = 0; i < 8; ++i) { afr[i] = Ap[k0 + i]; bfr[i] = Bp[k0 + i]; }
  };
  auto store_tiles = [&](int buf) {
    unsigned pa[4], pb[4];
#pragma unroll
    for (int i = 0; i < 8; i += 2) {
      float a0 = afr[i], a1 = afr[i + 1];
      if (A_AFF) {
        a0 = fmaxf(a0 * sA + oA, 0.0f);
        a1 = fmaxf(a1 * sA + oA, 0.0f);
      }
      pa[i >> 1] = packbf(a0, a1);
      pb[i >> 1] = packbf(bfr[i], bfr[i + 1]);
    }
    *(uint4*)(&As[buf][ar * LSTR + ak0]) = make_uint4(pa[0], pa[1], pa[2], pa[3]);
    *(uint4*)(&Bs[buf][ar * LSTR + ak0]) = make_uint4(pb[0], pb[1], pb[2], pb[3]);
  };
  auto compute = [&](int buf) {
    v16bf a  = load_a_frag(&As[buf][wm * 16 * LSTR], lane);
    v16bf bl = load_b_frag(&Bs[buf][(wn * 32)      * LSTR], lane);
    v16bf bh = load_b_frag(&Bs[buf][(wn * 32 + 16) * LSTR], lane);
    acc0 = wmma_bf16(a, bl, acc0);
    acc1 = wmma_bf16(a, bh, acc1);
  };

  const int nst = K >> 5;                    // K % 32 == 0
  fill_raw(0);
  store_tiles(0);
#pragma unroll 1
  for (int s = 1; s < nst; ++s) {
    __syncthreads();
    fill_raw(s * 32);
    compute((s - 1) & 1);
    store_tiles(s & 1);
  }
  __syncthreads();
  compute((nst - 1) & 1);

  const int ncol = lane & 15;
  const int mhi  = (lane & 16) ? 8 : 0;
  float s = 0.0f, q = 0.0f;
#pragma unroll
  for (int v = 0; v < 8; ++v) {
    int m = m0 + wm * 16 + v + mhi;
    int n = n0 + wn * 32 + ncol;
    if (OUT_BF16) {
      bf16 h0 = (bf16)acc0[v];
      bf16 h1 = (bf16)acc1[v];
      Cb[(size_t)m * N + n]      = h0;
      Cb[(size_t)m * N + n + 16] = h1;
      if (OSTATS) {   // stats on the rounded values actually consumed downstream
        float r0 = (float)h0;
        float r1 = (float)h1;
        s += r0 + r1; q += r0 * r0 + r1 * r1;
      }
    } else {
      Cf[(size_t)m * N + n]      = acc0[v];
      Cf[(size_t)m * N + n + 16] = acc1[v];
      if (OSTATS) {
        s += acc0[v] + acc1[v];
        q += acc0[v] * acc0[v] + acc1[v] * acc1[v];
      }
    }
  }
  if (OSTATS) {
    atomicAdd(&sblk, s); atomicAdd(&qblk, q);
    __syncthreads();
    if (tid == 0) {  // all 64 rows of this block share one (b, channel)
      int bb = m0 / 3072;
      int cc = (m0 >> 7) % 24;
      int g  = cc / 6;
      atomicAdd(&ostats[(bb * 4 + g) * 2 + 0], sblk);
      atomicAdd(&ostats[(bb * 4 + g) * 2 + 1], qblk);
    }
  }
}

// ---------------- GN + ReLU elementwise (for conv stages) ----------------
__global__ void __launch_bounds__(256)
gn_apply_relu(const float* __restrict__ y, const float* __restrict__ stats,
              const float* __restrict__ gw, const float* __restrict__ gb,
              float inv_count, float* __restrict__ o) {
  int idx = blockIdx.x * 256 + threadIdx.x;
  if (idx >= 6291456) return;
  int c  = (idx >> 16) % 24;
  int bb = idx / (24 * 65536);
  int g  = c / 6;
  float sum  = stats[(bb * 4 + g) * 2 + 0];
  float ssq  = stats[(bb * 4 + g) * 2 + 1];
  float mean = sum * inv_count;
  float var  = ssq * inv_count - mean * mean;
  float rs   = rsqrtf(var + GN_EPS);
  float v = (y[idx] - mean) * rs * gw[c] + gb[c];
  o[idx] = v > 0.0f ? v : 0.0f;
}

// ---------------- attention (q = c Wq^T + bq ; att = softmax(q keys / sqrt(dk))) ---
__global__ void att_kernel(const float* __restrict__ cvec, const float* __restrict__ wq,
                           const float* __restrict__ bq, const float* __restrict__ keys,
                           float* __restrict__ att) {
  int bb = threadIdx.x;
  if (bb >= 4) return;
  float q[32];
  for (int d = 0; d < 32; ++d) {
    float a = bq[d];
    for (int j = 0; j < 32; ++j) a += cvec[bb * 32 + j] * wq[d * 32 + j];
    q[d] = a;
  }
  float lg[6]; float mx = -1e30f;
  for (int n = 0; n < 6; ++n) {
    float a = 0.0f;
    for (int d = 0; d < 32; ++d) a += q[d] * keys[d * 6 + n];
    lg[n] = a * 0.17677669529663687f;        // 1/sqrt(32)
    mx = fmaxf(mx, lg[n]);
  }
  float se = 0.0f;
  for (int n = 0; n < 6; ++n) { lg[n] = __expf(lg[n] - mx); se += lg[n]; }
  for (int n = 0; n < 6; ++n) att[bb * 6 + n] = lg[n] / se;
}

// ---------------- final: out = x_ + sum_n relu(GN2(h2))[...,f,n] * att[b,n] --------
__global__ void __launch_bounds__(256)
final_mix(const float* __restrict__ xr, const unsigned* __restrict__ h2w,
          const float* __restrict__ stats, const float* __restrict__ gw,
          const float* __restrict__ gb, const float* __restrict__ att,
          float inv_count, float* __restrict__ out) {
  int idx = blockIdx.x * 256 + threadIdx.x;
  if (idx >= 6291456) return;
  int f   = idx & 511;
  int row = idx >> 9;                        // (b*24+c)*128 + t
  int c   = (row >> 7) % 24;
  int bb  = row / 3072;
  int g   = c / 6;
  float sum  = stats[(bb * 4 + g) * 2 + 0];
  float ssq  = stats[(bb * 4 + g) * 2 + 1];
  float mean = sum * inv_count;
  float var  = ssq * inv_count - mean * mean;
  float rs   = rsqrtf(var + GN_EPS);
  float sc = gw[c] * rs, of = gb[c] - mean * sc;
  const unsigned* p = h2w + (size_t)row * 1536 + f * 3;  // 6 bf16 = 3 dwords
  float acc = 0.0f;
#pragma unroll
  for (int w = 0; w < 3; ++w) {
    unsigned u = p[w];
    float v0 = __builtin_bit_cast(float, u << 16);
    float v1 = __builtin_bit_cast(float, u & 0xFFFF0000u);
    float a0 = v0 * sc + of; a0 = a0 > 0.0f ? a0 : 0.0f;
    float a1 = v1 * sc + of; a1 = a1 > 0.0f ? a1 : 0.0f;
    acc += a0 * att[bb * 6 + 2 * w] + a1 * att[bb * 6 + 2 * w + 1];
  }
  out[idx] = xr[idx] + acc;
}

__global__ void zero_stats(float* s, int n) {
  int i = blockIdx.x * 256 + threadIdx.x;
  if (i < n) s[i] = 0.0f;
}

// ---------------- launcher ----------------
extern "C" void kernel_launch(void* const* d_in, const int* in_sizes, int n_in,
                              void* d_out, int out_size, void* d_ws, size_t ws_size,
                              hipStream_t stream) {
  const float* x    = (const float*)d_in[0];
  const float* cv   = (const float*)d_in[1];
  const float* w0   = (const float*)d_in[2];
  const float* b0   = (const float*)d_in[3];
  const float* gw0  = (const float*)d_in[4];
  const float* gb0  = (const float*)d_in[5];
  const float* w1   = (const float*)d_in[6];
  const float* b1   = (const float*)d_in[7];
  const float* gw1  = (const float*)d_in[8];
  const float* gb1  = (const float*)d_in[9];
  const float* w2   = (const float*)d_in[10];
  const float* b2   = (const float*)d_in[11];
  const float* gw2  = (const float*)d_in[12];
  const float* gb2  = (const float*)d_in[13];
  const float* tw1  = (const float*)d_in[14];
  const float* tg1w = (const float*)d_in[15];
  const float* tg1b = (const float*)d_in[16];
  const float* tw2  = (const float*)d_in[17];
  const float* tg2w = (const float*)d_in[18];
  const float* tg2b = (const float*)d_in[19];
  const float* wq   = (const float*)d_in[20];
  const float* bq   = (const float*)d_in[21];
  const float* keys = (const float*)d_in[22];
  (void)in_sizes; (void)n_in; (void)out_size; (void)ws_size;

  char* ws = (char*)d_ws;
  size_t off = 0;
  auto alloc = [&](size_t bytes) { void* p = ws + off; off += (bytes + 255) & ~(size_t)255; return p; };
  float* stats = (float*)alloc(5 * 16 * 2 * sizeof(float));   // 5 GN stat slots
  float* att   = (float*)alloc(4 * 6 * sizeof(float));
  float* yraw  = (float*)alloc(6291456ull * 4);               // conv out (pre-GN), reused
  float* feat0 = (float*)alloc(6291456ull * 4);
  float* feat1 = (float*)alloc(6291456ull * 4);
  float* xund  = (float*)alloc(6291456ull * 4);               // x_ (post conv stack)
  float* hraw  = (float*)alloc(2359296ull * 4);               // h pre-GN [12288,192]
  bf16*  h2    = (bf16*) alloc(37748736ull * 2);              // h2 pre-GN [12288,3072] bf16
  float* out   = (float*)d_out;

  zero_stats<<<1, 256, 0, stream>>>(stats, 160);

  const dim3 cgrid(8, 128, 4);   // f-tiles, T, B
  const float ic_conv = 1.0f / 393216.0f;   // 6*128*512

  // conv stage 0: in = x (24ch); GN stats fused into epilogue
  conv3x3_wmma<24><<<cgrid, 128, 0, stream>>>(x, nullptr, nullptr, w0, b0, yraw, stats + 0);
  gn_apply_relu<<<24576, 256, 0, stream>>>(yraw, stats + 0, gw0, gb0, ic_conv, feat0);

  // conv stage 1: in = [feat0, x] (48ch, concat read virtually)
  conv3x3_wmma<48><<<cgrid, 128, 0, stream>>>(feat0, x, nullptr, w1, b1, yraw, stats + 32);
  gn_apply_relu<<<24576, 256, 0, stream>>>(yraw, stats + 32, gw1, gb1, ic_conv, feat1);

  // conv stage 2: in = [feat1, feat0, x] (72ch)
  conv3x3_wmma<72><<<cgrid, 128, 0, stream>>>(feat1, feat0, x, w2, b2, yraw, stats + 64);
  gn_apply_relu<<<24576, 256, 0, stream>>>(yraw, stats + 64, gw2, gb2, ic_conv, xund);

  // TDF GEMM1: h = x_ [12288,512] * w1^T [512,192] ; GN1 stats fused
  gemm_wmma<false, false, true><<<dim3(192, 3), 256, 0, stream>>>(
      xund, tw1, hraw, nullptr, 12288, 192, 512,
      nullptr, nullptr, nullptr, 0.0f, stats + 96);

  // TDF GEMM2: GN1+ReLU fused into A load; bf16 out; GN2 stats fused
  gemm_wmma<true, true, true><<<dim3(192, 48), 256, 0, stream>>>(
      hraw, tw2, nullptr, h2, 12288, 3072, 192,
      stats + 96, tg1w, tg1b, 1.0f / 147456.0f, stats + 128);

  att_kernel<<<1, 32, 0, stream>>>(cv, wq, bq, keys, att);

  // out = x_ + sum_n relu(GN2(h2)) * att   (GN2 fused here)
  final_mix<<<24576, 256, 0, stream>>>(xund, (const unsigned*)h2, stats + 128,
                                       tg2w, tg2b, att, 1.0f / 2359296.0f, out);
}
